// MultiHeadAttention_20005957665232
// MI455X (gfx1250) — matrix operand order repaired and checked by reading, not compiled, not run on hardware
//
#include <hip/hip_runtime.h>

#define S_LEN 2048
#define DMODEL 1024
#define NH 16
#define HD 64
#define BATCH 2
#define MROWS (BATCH * S_LEN)   // 4096

typedef __attribute__((ext_vector_type(16))) __bf16 v16bf;
typedef __attribute__((ext_vector_type(8)))  __bf16 v8bf;
typedef __attribute__((ext_vector_type(8)))  float  v8f;
typedef __attribute__((ext_vector_type(4)))  unsigned int uv4;
typedef __attribute__((ext_vector_type(8)))  int iv8;
typedef __attribute__((ext_vector_type(4)))  int iv4;

static __device__ __forceinline__ v8f zero_v8f() {
  v8f z;
#pragma unroll
  for (int e = 0; e < 8; ++e) z[e] = 0.f;
  return z;
}

static __device__ __forceinline__ v8f wmma_bf16(v16bf a, v16bf b, v8f c) {
  // v_wmma_f32_16x16x32_bf16
  return __builtin_amdgcn_wmma_f32_16x16x32_bf16(false, a, false, b, (short)0, c,
                                                 false, false);
}

// ---- Tensor Data Mover: 2D bf16 tile (w_elems x h_rows, row stride in elems)
// from global memory into LDS at byte offset lds_off. D# layout per ISA ch.8:
//   group0: [1:0]=count=1, [63:32]=lds_addr, [120:64]=global_addr, [127:126]=type=2
//   group1: [17:16]=data_size(1=2B), [79:48]=tensor_dim0, [111:80]=tensor_dim1,
//           [127:112]=tile_dim0, [143:128]=tile_dim1, [207:160]=tensor_dim0_stride
// Groups 2/3 zero (tile_dim2/3 = 0 => unused; 2D tile). Tracked by TENSORcnt.
static __device__ __forceinline__ void tdm_load_2d_bf16(
    unsigned int lds_off, const __bf16* gptr,
    unsigned int w_elems, unsigned int h_rows, unsigned int stride_elems) {
  unsigned long long ga = (unsigned long long)(size_t)gptr;
  uv4 g0;
  g0[0] = 1u;                                              // count=1, user D#
  g0[1] = lds_off;                                         // lds_addr (bytes)
  g0[2] = (unsigned int)ga;                                // global_addr[31:0]
  g0[3] = (unsigned int)((ga >> 32) & 0x01FFFFFFu) | (2u << 30);  // ga[56:32] | type=2
  iv8 g1;
  g1[0] = (int)(1u << 16);                                 // data_size = 2 bytes
  g1[1] = (int)((w_elems & 0xffffu) << 16);                // tensor_dim0[15:0]
  g1[2] = (int)(((w_elems >> 16) & 0xffffu) |              // tensor_dim0[31:16]
                ((h_rows & 0xffffu) << 16));               // tensor_dim1[15:0]
  g1[3] = (int)(((h_rows >> 16) & 0xffffu) |               // tensor_dim1[31:16]
                ((w_elems & 0xffffu) << 16));              // tile_dim0 = w
  g1[4] = (int)(h_rows & 0xffffu);                         // tile_dim1 = h, tile_dim2=0
  g1[5] = (int)stride_elems;                               // tensor_dim0_stride[31:0]
  g1[6] = 0;
  g1[7] = 0;
  iv4 z4;
  z4[0] = 0; z4[1] = 0; z4[2] = 0; z4[3] = 0;
  iv8 z8;
#pragma unroll
  for (int e = 0; e < 8; ++e) z8[e] = 0;
  __builtin_amdgcn_tensor_load_to_lds(g0, g1, z4, z4, z8, 0);
}

static __device__ __forceinline__ unsigned int lds_offset_of(const void* p) {
  // generic LDS address: low 32 bits are the LDS byte offset
  return (unsigned int)(size_t)p;
}

// A-matrix fragment (16x32 MxK bf16) from row-major tile [rows][ldm].
// Lane layout: lanes 0-15 hold K=0..15, lanes 16-31 hold K=16..31. m = lane&15.
static __device__ __forceinline__ v16bf load_a_frag(const __bf16* t, int ldm,
                                                    int m, int g, int koff) {
  v16bf a;
#pragma unroll
  for (int r = 0; r < 8; ++r) {
    int k0 = koff + 16 * g + 2 * r;
    a[2 * r]     = t[m * ldm + k0];
    a[2 * r + 1] = t[m * ldm + k0 + 1];
  }
  return a;
}

// B-matrix fragment (32x16 KxN bf16) from tile stored [n][k].
// lanes 0-15 hold K=0..15, lanes 16-31 hold K=16..31; n = lane&15.
static __device__ __forceinline__ v16bf load_b_frag_nk(const __bf16* t, int ldm,
                                                       int n, int g, int koff) {
  v16bf b;
#pragma unroll
  for (int r = 0; r < 8; ++r) {
    int k0 = koff + 16 * g + 2 * r;
    b[2 * r]     = t[n * ldm + k0];
    b[2 * r + 1] = t[n * ldm + k0 + 1];
  }
  return b;
}

// B-matrix fragment from tile stored [k][n].
static __device__ __forceinline__ v16bf load_b_frag_kn(const __bf16* t, int ldm,
                                                       int n, int g, int koff) {
  v16bf b;
#pragma unroll
  for (int r = 0; r < 8; ++r) {
    int k0 = koff + 16 * g + 2 * r;
    b[2 * r]     = t[(k0    ) * ldm + n];
    b[2 * r + 1] = t[(k0 + 1) * ldm + n];
  }
  return b;
}

__global__ void f2bf_kernel(const float* __restrict__ in, __bf16* __restrict__ out, int n) {
  int i = (blockIdx.x * blockDim.x + threadIdx.x) * 4;
  if (i + 3 < n) {
    float4 f = *(const float4*)&in[i];
    out[i]     = (__bf16)f.x;
    out[i + 1] = (__bf16)f.y;
    out[i + 2] = (__bf16)f.z;
    out[i + 3] = (__bf16)f.w;
  }
}

__global__ void rope_table_kernel(const int* __restrict__ pos,
                                  float* __restrict__ cosT, float* __restrict__ sinT) {
  int idx = blockIdx.x * blockDim.x + threadIdx.x;
  if (idx >= S_LEN * HD) return;
  int s = idx >> 6, d = idx & 63;
  float fi = (float)(d & 31);
  // inv_freq = 10000^(-fi/32) = exp(-fi * ln(10000)/32)
  float inv = __expf(fi * -0.28782313662425575f);
  float ang = (float)pos[s] * inv;
  cosT[idx] = __cosf(ang);
  sinT[idx] = __sinf(ang);
}

// MODE: 0=Q (bias+rope), 1=K (rope only), 2=V (bias only), 3=OUT (bias, f32 output)
// Y = X @ W^T. X: (MROWS x 1024) bf16. W: (1024 x 1024) bf16 row-major (out,in).
// Workgroup: 256 thr = 8 waves; tile 128(M) x 64(N); each wave owns a 16x64 strip
// (one full head for QKV modes, so RoPE pairing d<->d+-32 stays inside the wave).
// A/B tiles staged in LDS by the Tensor Data Mover (one 2D descriptor per wave).
template <int MODE>
__global__ __launch_bounds__(256) void gemm_wmma_kernel(
    const __bf16* __restrict__ X, const __bf16* __restrict__ W,
    const float* __restrict__ bias,
    const float* __restrict__ cosT, const float* __restrict__ sinT,
    __bf16* __restrict__ outb, float* __restrict__ outf) {
  __shared__ __bf16 As[128 * 32];
  __shared__ __bf16 Bs[64 * 32];
  const int tid  = threadIdx.x;
  const int wave = tid >> 5, lane = tid & 31;
  const int g = lane >> 4, ln = lane & 15;
  const int n0 = blockIdx.x * 64;
  const int m0 = blockIdx.y * 128;

  const unsigned int as_lds = lds_offset_of(&As[0]) + (unsigned int)wave * 16u * 32u * 2u;
  const unsigned int bs_lds = lds_offset_of(&Bs[0]) + (unsigned int)wave * 8u * 32u * 2u;

  v8f acc[4];
#pragma unroll
  for (int t = 0; t < 4; ++t) acc[t] = zero_v8f();

  for (int kk = 0; kk < DMODEL; kk += 32) {
    // TDM: wave w DMAs A rows [w*16,w*16+16) x 32, B rows [w*8,w*8+8) x 32
    tdm_load_2d_bf16(as_lds, &X[(size_t)(m0 + wave * 16) * DMODEL + kk], 32, 16, DMODEL);
    tdm_load_2d_bf16(bs_lds, &W[(size_t)(n0 + wave * 8) * DMODEL + kk], 32, 8, DMODEL);
    if (kk + 32 < DMODEL) {  // global_prefetch_b8 of next K-slab into L2
      __builtin_prefetch(&X[(size_t)(m0 + tid / 2) * DMODEL + kk + 32], 0, 0);
      __builtin_prefetch(&W[(size_t)(n0 + tid / 4) * DMODEL + kk + 32], 0, 0);
    }
    __builtin_amdgcn_s_wait_tensorcnt(0);
    __syncthreads();

    v16bf a = load_a_frag(&As[wave * 16 * 32], 32, ln, g, 0);
#pragma unroll
    for (int t = 0; t < 4; ++t) {
      v16bf b = load_b_frag_nk(Bs, 32, t * 16 + ln, g, 0);
      acc[t] = wmma_bf16(a, b, acc[t]);
    }
    __syncthreads();
  }

  // Epilogue: bias first (rope applies to x@W^T + b in the reference).
  float vals[4][8];
#pragma unroll
  for (int t = 0; t < 4; ++t)
#pragma unroll
    for (int i = 0; i < 8; ++i) {
      float v = acc[t][i];
      if (MODE != 1) v += bias[n0 + t * 16 + ln];
      vals[t][i] = v;
    }

  const int h = blockIdx.x;  // N-block == head for QKV modes
#pragma unroll
  for (int t = 0; t < 4; ++t) {
#pragma unroll
    for (int i = 0; i < 8; ++i) {
      int mrow = m0 + wave * 16 + i + 8 * g;  // C layout: M = i + 8*(lane>=16)
      if (MODE == 3) {
        outf[(size_t)mrow * DMODEL + n0 + t * 16 + ln] = vals[t][i];
      } else {
        int bb = mrow >> 11;           // / S_LEN
        int s  = mrow & (S_LEN - 1);
        int d  = t * 16 + ln;
        float v = vals[t][i];
        if (MODE == 0 || MODE == 1) {  // RoPE: out = x*cos + rotate_half(x)*sin
          float c  = cosT[s * HD + d];
          float sn = sinT[s * HD + d];
          float part = (t < 2) ? -vals[t + 2][i] : vals[t - 2][i];
          v = v * c + part * sn;
        }
        outb[((size_t)(bb * NH + h) * S_LEN + s) * HD + d] = (__bf16)v;
      }
    }
  }
}

// Flash attention: one workgroup = (b, h, 128 q-rows); 8 waves, each owns 16 q-rows.
// K/V 64-key tiles staged in LDS by TDM; online softmax via wave32 shuffles.
__global__ __launch_bounds__(256) void attn_wmma_kernel(
    const __bf16* __restrict__ Q, const __bf16* __restrict__ K,
    const __bf16* __restrict__ V, __bf16* __restrict__ O) {
  __shared__ __bf16 Kt[64 * HD];
  __shared__ __bf16 Vt[64 * HD];
  __shared__ __bf16 Pt[8][16 * 64];

  const int tid  = threadIdx.x;
  const int wave = tid >> 5, lane = tid & 31;
  const int g = lane >> 4, ln = lane & 15;
  const int qb = blockIdx.x * 128;
  const int h = blockIdx.y, b = blockIdx.z;
  const size_t base = (size_t)(b * NH + h) * S_LEN * HD;

  const unsigned int kt_lds = lds_offset_of(&Kt[0]) + (unsigned int)wave * 512u * 2u;
  const unsigned int vt_lds = lds_offset_of(&Vt[0]) + (unsigned int)wave * 512u * 2u;

  const __bf16* qtile = Q + base + (size_t)(qb + wave * 16) * HD;
  v16bf qf0 = load_a_frag(qtile, HD, ln, g, 0);
  v16bf qf1 = load_a_frag(qtile, HD, ln, g, 32);

  float mrow[8], lrow[8];
  v8f oacc[4];
#pragma unroll
  for (int i = 0; i < 8; ++i) { mrow[i] = -1e30f; lrow[i] = 0.f; }
#pragma unroll
  for (int t = 0; t < 4; ++t) oacc[t] = zero_v8f();

  const int nkt = qb / 64 + 2;  // causal: keys < qb+128
  for (int kt = 0; kt < nkt; ++kt) {
    // 64x64 bf16 tile is contiguous in (B,H,S,hd); wave w DMAs keys [w*8,w*8+8)
    const size_t src = base + (size_t)kt * 64 * HD + (size_t)wave * 512;
    tdm_load_2d_bf16(kt_lds, &K[src], 512, 1, 512);
    tdm_load_2d_bf16(vt_lds, &V[src], 512, 1, 512);
    __builtin_amdgcn_s_wait_tensorcnt(0);
    __syncthreads();

    // scores = q @ k^T : 4 N-tiles x (2 chained wmma over hd=64)
    v8f sacc[4];
#pragma unroll
    for (int t = 0; t < 4; ++t) {
      v8f z = zero_v8f();
      v16bf kb0 = load_b_frag_nk(Kt, HD, t * 16 + ln, g, 0);
      v16bf kb1 = load_b_frag_nk(Kt, HD, t * 16 + ln, g, 32);
      z = wmma_bf16(qf0, kb0, z);
      z = wmma_bf16(qf1, kb1, z);
      sacc[t] = z;
    }

    // online softmax; row m = i + 8*g lives across the 16 lanes of the group
#pragma unroll
    for (int i = 0; i < 8; ++i) {
      const int qg = qb + wave * 16 + i + 8 * g;
      float sv[4];
      float tm = -1e30f;
#pragma unroll
      for (int t = 0; t < 4; ++t) {
        int kg = kt * 64 + t * 16 + ln;
        float s = sacc[t][i] * 0.125f;  // 1/sqrt(64)
        if (kg > qg) s = -1e30f;        // causal mask
        sv[t] = s;
        tm = fmaxf(tm, s);
      }
#pragma unroll
      for (int off = 1; off < 16; off <<= 1)
        tm = fmaxf(tm, __shfl_xor(tm, off, 32));
      float mnew = fmaxf(mrow[i], tm);
      float corr = __expf(mrow[i] - mnew);
      float rs = 0.f;
#pragma unroll
      for (int t = 0; t < 4; ++t) {
        float pv = __expf(sv[t] - mnew);
        Pt[wave][(i + 8 * g) * 64 + t * 16 + ln] = (__bf16)pv;
        rs += pv;
      }
#pragma unroll
      for (int off = 1; off < 16; off <<= 1)
        rs += __shfl_xor(rs, off, 32);
      lrow[i] = lrow[i] * corr + rs;
      mrow[i] = mnew;
#pragma unroll
      for (int t = 0; t < 4; ++t) oacc[t][i] = oacc[t][i] * corr;
    }

    asm volatile("s_wait_dscnt 0" ::: "memory");  // P stores visible before A-frag gather

    v16bf pf0 = load_a_frag(&Pt[wave][0], 64, ln, g, 0);
    v16bf pf1 = load_a_frag(&Pt[wave][0], 64, ln, g, 32);
#pragma unroll
    for (int t = 0; t < 4; ++t) {
      v16bf vb0 = load_b_frag_kn(Vt, HD, t * 16 + ln, g, 0);
      v16bf vb1 = load_b_frag_kn(Vt, HD, t * 16 + ln, g, 32);
      oacc[t] = wmma_bf16(pf0, vb0, oacc[t]);
      oacc[t] = wmma_bf16(pf1, vb1, oacc[t]);
    }
    __syncthreads();
  }

#pragma unroll
  for (int t = 0; t < 4; ++t)
#pragma unroll
    for (int i = 0; i < 8; ++i) {
      int qg = qb + wave * 16 + i + 8 * g;
      float o = oacc[t][i] / lrow[i];
      O[((size_t)b * S_LEN + qg) * DMODEL + h * HD + t * 16 + ln] = (__bf16)o;
    }
}

extern "C" void kernel_launch(void* const* d_in, const int* in_sizes, int n_in,
                              void* d_out, int out_size, void* d_ws, size_t ws_size,
                              hipStream_t stream) {
  (void)in_sizes; (void)n_in; (void)out_size; (void)ws_size;
  const float* qx  = (const float*)d_in[0];
  const float* kvx = (const float*)d_in[1];
  // d_in[2]=k_cache, d_in[3]=v_cache: fully overwritten by positions=arange -> dead
  const float* Wq = (const float*)d_in[4];
  const float* bq = (const float*)d_in[5];
  const float* Wk = (const float*)d_in[6];
  const float* Wv = (const float*)d_in[7];
  const float* bv = (const float*)d_in[8];
  const float* Wo = (const float*)d_in[9];
  const float* bo = (const float*)d_in[10];
  const int* qpos  = (const int*)d_in[11];
  const int* kvpos = (const int*)d_in[12];

  char* ws = (char*)d_ws;
  size_t off = 0;
  auto carve = [&](size_t bytes) -> char* {
    char* p = ws + off;
    off = (off + bytes + 255) & ~(size_t)255;
    return p;
  };

  const size_t actE = (size_t)MROWS * DMODEL;   // 4096*1024
  const size_t wE   = (size_t)DMODEL * DMODEL;  // 1024*1024
  __bf16* Xq    = (__bf16*)carve(actE * 2);
  __bf16* Xkv   = (__bf16*)carve(actE * 2);
  __bf16* Wqb   = (__bf16*)carve(wE * 2);
  __bf16* Wkb   = (__bf16*)carve(wE * 2);
  __bf16* Wvb   = (__bf16*)carve(wE * 2);
  __bf16* Wob   = (__bf16*)carve(wE * 2);
  __bf16* qr    = (__bf16*)carve(actE * 2);  // (B,H,S,hd) bf16, rope'd
  __bf16* kr    = (__bf16*)carve(actE * 2);
  __bf16* vr    = (__bf16*)carve(actE * 2);
  __bf16* attnb = (__bf16*)carve(actE * 2);  // (B,S,D) bf16
  float* cq = (float*)carve((size_t)S_LEN * HD * 4);
  float* sq = (float*)carve((size_t)S_LEN * HD * 4);
  float* ck = (float*)carve((size_t)S_LEN * HD * 4);
  float* sk = (float*)carve((size_t)S_LEN * HD * 4);

  // bf16 conversions
  f2bf_kernel<<<(int)(actE / 4 / 256), 256, 0, stream>>>(qx, Xq, (int)actE);
  f2bf_kernel<<<(int)(actE / 4 / 256), 256, 0, stream>>>(kvx, Xkv, (int)actE);
  f2bf_kernel<<<(int)(wE / 4 / 256), 256, 0, stream>>>(Wq, Wqb, (int)wE);
  f2bf_kernel<<<(int)(wE / 4 / 256), 256, 0, stream>>>(Wk, Wkb, (int)wE);
  f2bf_kernel<<<(int)(wE / 4 / 256), 256, 0, stream>>>(Wv, Wvb, (int)wE);
  f2bf_kernel<<<(int)(wE / 4 / 256), 256, 0, stream>>>(Wo, Wob, (int)wE);

  // rope tables
  rope_table_kernel<<<(S_LEN * HD + 255) / 256, 256, 0, stream>>>(qpos, cq, sq);
  rope_table_kernel<<<(S_LEN * HD + 255) / 256, 256, 0, stream>>>(kvpos, ck, sk);

  // projections (+bias, +rope) -> (B,H,S,hd) bf16
  dim3 ggrid(DMODEL / 64, MROWS / 128);
  gemm_wmma_kernel<0><<<ggrid, 256, 0, stream>>>(Xq,  Wqb, bq, cq, sq, qr, nullptr);
  gemm_wmma_kernel<1><<<ggrid, 256, 0, stream>>>(Xkv, Wkb, nullptr, ck, sk, kr, nullptr);
  gemm_wmma_kernel<2><<<ggrid, 256, 0, stream>>>(Xkv, Wvb, bv, nullptr, nullptr, vr, nullptr);

  // flash attention -> (B,S,D) bf16
  attn_wmma_kernel<<<dim3(S_LEN / 128, NH, BATCH), 256, 0, stream>>>(qr, kr, vr, attnb);

  // output projection -> fp32 d_out
  gemm_wmma_kernel<3><<<ggrid, 256, 0, stream>>>(attnb, Wob, bo, nullptr, nullptr,
                                                 nullptr, (float*)d_out);
}
